// GCN_REGR_618475290682
// MI455X (gfx1250) — compile-verified, matrix operand order checked by reading, not run on hardware
//
#include <hip/hip_runtime.h>

#define NN   100000   // nodes  (multiple of 16 -> no M tail in WMMA kernel)
#define NE   1600000  // edges
#define NG   256      // graphs
#define HID  40       // hidden dim
#define KPAD 64       // padded K for f16 staging (covers K=30 and K=40)

typedef __attribute__((ext_vector_type(16))) _Float16 v16h;
typedef __attribute__((ext_vector_type(8)))  _Float16 v8h;
typedef __attribute__((ext_vector_type(8)))  float    v8f;

// ---------------- init / degree ----------------

__global__ void k_init(float* __restrict__ deg, float* __restrict__ gsum,
                       float* __restrict__ gcnt) {
  int i = blockIdx.x * blockDim.x + threadIdx.x;
  if (i < NN) deg[i] = 1.0f;            // self loop contributes 1 to degree
  if (i < NG * HID) gsum[i] = 0.0f;
  if (i < NG) gcnt[i] = 0.0f;
}

__global__ void k_zero(float* __restrict__ p, int n) {
  int i = blockIdx.x * blockDim.x + threadIdx.x;
  if (i < n) p[i] = 0.0f;
}

__global__ void k_degree(const int* __restrict__ dst, float* __restrict__ deg) {
  int e = blockIdx.x * blockDim.x + threadIdx.x;
  if (e < NE) atomicAdd(&deg[dst[e]], 1.0f);
}

__global__ void k_dinv(float* __restrict__ deg) {
  int i = blockIdx.x * blockDim.x + threadIdx.x;
  if (i < NN) deg[i] = rsqrtf(fmaxf(deg[i], 1.0f));   // in-place: deg -> dinv
}

// ---------------- operand packing (f32 -> padded f16) ----------------

// in [NN, kdim] f32  ->  outp [NN, KPAD] f16, zero padded, k-major rows (128 B)
__global__ void k_pack_x(const float* __restrict__ in, int kdim,
                         _Float16* __restrict__ outp) {
  int i = blockIdx.x * blockDim.x + threadIdx.x;     // over NN*KPAD
  if (i >= NN * KPAD) return;
  int node = i >> 6, k = i & (KPAD - 1);
  outp[i] = (k < kdim) ? (_Float16)in[node * kdim + k] : (_Float16)0.0f;
}

// W [kdim, HID] f32 -> B fragments in per-lane element order:
// outp[((kc*3 + t)*32 + lane)*16 + j], mapping (ISA 7.12.2, transposed for B):
//   khi = (lane>=16)*8 ; k = kc*32 + (j/8)*16 + khi + (j%8) ; col = t*16 + lane%16
__global__ void k_pack_w(const float* __restrict__ W, int kdim, int kchunks,
                         _Float16* __restrict__ outp) {
  int i = blockIdx.x * blockDim.x + threadIdx.x;     // over kchunks*3*32*16
  if (i >= kchunks * 3 * 32 * 16) return;
  int j    = i & 15;
  int lane = (i >> 4) & 31;
  int frag = i >> 9;                                  // kc*3 + t
  int kc = frag / 3, t = frag % 3;
  int khi = (lane >> 4) * 8;
  int k   = kc * 32 + ((j >> 3) << 4) + khi + (j & 7);
  int col = t * 16 + (lane & 15);
  outp[i] = (k < kdim && col < HID) ? (_Float16)W[k * HID + col] : (_Float16)0.0f;
}

// ---------------- dense transform via WMMA ----------------
// out[m, n] = sum_k in[m, k] * W[k, n]; one wave = 16-row x 48-col tile.
// A fragment: two aligned b128 loads from the packed 128-B row.
// B fragment: two contiguous b128 loads from the pre-packed fragment buffer.
template<int KCHUNKS>
__global__ void k_dense_wmma(const _Float16* __restrict__ Apack,  // [NN][KPAD]
                             const _Float16* __restrict__ Bpack,  // [KCHUNKS*3][32][16]
                             float* __restrict__ out) {            // [NN][HID]
  const int lane = threadIdx.x & 31;
  const int wave = blockIdx.x * (blockDim.x >> 5) + (threadIdx.x >> 5);
  const int m0 = wave * 16;
  if (m0 >= NN) return;                 // whole wave exits: EXEC stays all-1s

  const int l16 = lane & 15;
  const int khi = (lane >> 4) * 8;
  const _Float16* arow = Apack + (m0 + l16) * KPAD;

  v8f acc[3] = {};
#pragma unroll
  for (int kc = 0; kc < KCHUNKS; ++kc) {
    union { v16h v; v8h h[2]; } a;
    a.h[0] = *(const v8h*)(arow + kc * 32 + khi);        // K = kc*32 + khi .. +7
    a.h[1] = *(const v8h*)(arow + kc * 32 + 16 + khi);   // K = kc*32 + 16+khi .. +7
#pragma unroll
    for (int t = 0; t < 3; ++t) {
      const _Float16* bp = Bpack + (((kc * 3 + t) * 32) + lane) * 16;
      union { v16h v; v8h h[2]; } b;
      b.h[0] = *(const v8h*)(bp);
      b.h[1] = *(const v8h*)(bp + 8);
      acc[t] = __builtin_amdgcn_wmma_f32_16x16x32_f16(
          false, a.v, false, b.v, (short)0, acc[t], false, false);
    }
  }

  // C/D layout: VGPR i -> M = m0 + 8*(lane>=16) + i, N = t*16 + lane%16
  const int rbase = m0 + (lane >> 4) * 8;
#pragma unroll
  for (int t = 0; t < 3; ++t) {
    const int col = t * 16 + l16;
    if (t < 2 || l16 < 8) {             // tile 2 covers cols 32..47; only 32..39 valid
#pragma unroll
      for (int i = 0; i < 8; ++i)
        out[(rbase + i) * HID + col] = acc[t][i];
    }
  }
}

// ---------------- edge scatter: accum[dst] += dinv[s]*dinv[d] * h[src] ----------------
// gridDim.y = 10 feature groups of 4; float4 gather keeps each access 16B (L2-resident).
__global__ void k_scatter(const float* __restrict__ h, const int* __restrict__ src,
                          const int* __restrict__ dst, const float* __restrict__ dinv,
                          float* __restrict__ accum) {
  const int e = blockIdx.x * blockDim.x + threadIdx.x;
  if (e >= NE) return;
  const int fg = blockIdx.y;            // 0..9
  const int s = src[e], d = dst[e];
  const float w = dinv[s] * dinv[d];
  const float4 hv = *(const float4*)(h + s * HID + fg * 4);
  float* base = accum + d * HID + fg * 4;
  atomicAdd(base + 0, w * hv.x);
  atomicAdd(base + 1, w * hv.y);
  atomicAdd(base + 2, w * hv.z);
  atomicAdd(base + 3, w * hv.w);
}

// out = relu(accum + dinv^2 * dense + bias)   (self-loop term folded analytically)
__global__ void k_finalize(const float* __restrict__ accum, const float* __restrict__ dense,
                           const float* __restrict__ dinv, const float* __restrict__ bias,
                           float* __restrict__ out) {
  const int i = blockIdx.x * blockDim.x + threadIdx.x;
  if (i >= NN * HID) return;
  const int node = i / HID, f = i % HID;
  const float di = dinv[node];
  const float v = accum[i] + di * di * dense[i] + bias[f];
  out[i] = fmaxf(v, 0.0f);
}

// ---------------- mean pool + linear head ----------------

__global__ void k_pool(const float* __restrict__ h, const int* __restrict__ batch,
                       float* __restrict__ gsum, float* __restrict__ gcnt) {
  const int i = blockIdx.x * blockDim.x + threadIdx.x;
  if (i >= NN * HID) return;
  const int node = i / HID, f = i % HID;
  const int g = batch[node];
  atomicAdd(&gsum[g * HID + f], h[i]);
  if (f == 0) atomicAdd(&gcnt[g], 1.0f);
}

__global__ void k_head(const float* __restrict__ gsum, const float* __restrict__ gcnt,
                       const float* __restrict__ lw, const float* __restrict__ lb,
                       float* __restrict__ out) {
  const int g = blockIdx.x * blockDim.x + threadIdx.x;
  if (g >= NG) return;
  const float inv = 1.0f / fmaxf(gcnt[g], 1.0f);
  float acc = lb[0];
#pragma unroll
  for (int f = 0; f < HID; ++f) acc += gsum[g * HID + f] * inv * lw[f];
  out[g] = acc;
}

// ---------------- launch ----------------

extern "C" void kernel_launch(void* const* d_in, const int* in_sizes, int n_in,
                              void* d_out, int out_size, void* d_ws, size_t ws_size,
                              hipStream_t stream) {
  (void)in_sizes; (void)n_in; (void)out_size; (void)ws_size;

  const float* x  = (const float*)d_in[0];
  const int*  ei  = (const int*)d_in[1];
  const int*  src = ei;
  const int*  dst = ei + NE;
  const int*  bat = (const int*)d_in[2];
  const float* W1 = (const float*)d_in[3];  const float* b1 = (const float*)d_in[4];
  const float* W2 = (const float*)d_in[5];  const float* b2 = (const float*)d_in[6];
  const float* W3 = (const float*)d_in[7];  const float* b3 = (const float*)d_in[8];
  const float* lw = (const float*)d_in[9];  const float* lb = (const float*)d_in[10];
  float* out = (float*)d_out;

  // workspace layout; all sections stay 16B-aligned
  float* ws   = (float*)d_ws;
  float* dinv = ws;                       // NN floats
  float* hA   = ws + NN;                  // NN*HID (layer input / final features)
  float* hB   = hA + NN * HID;            // NN*HID (dense output)
  float* hC   = hB + NN * HID;            // NN*HID (scatter accumulator)
  float* gsum = hC + NN * HID;            // NG*HID
  float* gcnt = gsum + NG * HID;          // NG
  _Float16* xpack = (_Float16*)(gcnt + NG);          // NN*KPAD halves (12.8 MB)
  _Float16* wpack = xpack + (size_t)NN * KPAD;       // 2*3*32*16 halves

  const int B = 256;
  const dim3 gN((NN + B - 1) / B);              // 391
  const dim3 gE((NE + B - 1) / B);              // 6250
  const dim3 gF((NN * HID + B - 1) / B);        // 15625
  const dim3 gP((NN * KPAD + B - 1) / B);       // 25000
  const int wavesPerBlock = B / 32;             // 8
  const dim3 gW((NN / 16 + wavesPerBlock - 1) / wavesPerBlock);  // 782
  const dim3 gS((NE + B - 1) / B, 10);          // edge x feature-group

  k_init<<<gN, B, 0, stream>>>(dinv, gsum, gcnt);
  k_degree<<<gE, B, 0, stream>>>(dst, dinv);
  k_dinv<<<gN, B, 0, stream>>>(dinv);

  // ----- layer 1: x[NN,30] @ W1[30,40] -----
  k_pack_x<<<gP, B, 0, stream>>>(x, 30, xpack);
  k_pack_w<<<dim3(6), B, 0, stream>>>(W1, 30, 1, wpack);
  k_dense_wmma<1><<<gW, B, 0, stream>>>(xpack, wpack, hB);
  k_zero<<<gF, B, 0, stream>>>(hC, NN * HID);
  k_scatter<<<gS, B, 0, stream>>>(hB, src, dst, dinv, hC);
  k_finalize<<<gF, B, 0, stream>>>(hC, hB, dinv, b1, hA);

  // ----- layer 2: hA[NN,40] @ W2[40,40] -----
  k_pack_x<<<gP, B, 0, stream>>>(hA, HID, xpack);
  k_pack_w<<<dim3(12), B, 0, stream>>>(W2, HID, 2, wpack);
  k_dense_wmma<2><<<gW, B, 0, stream>>>(xpack, wpack, hB);
  k_zero<<<gF, B, 0, stream>>>(hC, NN * HID);
  k_scatter<<<gS, B, 0, stream>>>(hB, src, dst, dinv, hC);
  k_finalize<<<gF, B, 0, stream>>>(hC, hB, dinv, b2, hA);

  // ----- layer 3: hA[NN,40] @ W3[40,40] -----
  k_pack_x<<<gP, B, 0, stream>>>(hA, HID, xpack);
  k_pack_w<<<dim3(12), B, 0, stream>>>(W3, HID, 2, wpack);
  k_dense_wmma<2><<<gW, B, 0, stream>>>(xpack, wpack, hB);
  k_zero<<<gF, B, 0, stream>>>(hC, NN * HID);
  k_scatter<<<gS, B, 0, stream>>>(hB, src, dst, dinv, hC);
  k_finalize<<<gF, B, 0, stream>>>(hC, hB, dinv, b3, hA);

  // ----- mean pool + head -----
  k_pool<<<gF, B, 0, stream>>>(hA, bat, gsum, gcnt);
  k_head<<<dim3(1), B, 0, stream>>>(gsum, gcnt, lw, lb, out);
}
